// TokenChoiceMoE_85109071937953
// MI455X (gfx1250) — matrix operand order repaired and checked by reading, not compiled, not run on hardware
//
#include <hip/hip_runtime.h>
#include <hip/hip_bf16.h>
#include <math.h>

// ---- problem constants (from reference setup_inputs) ----
#define BATCH 2
#define LEN   2048
#define NTOK  (BATCH*LEN)     // 4096
#define DIM   1024
#define NEXP  8
#define FDIM  4096
#define TOPK  2

typedef __attribute__((ext_vector_type(16))) __bf16 v16bf;
typedef __attribute__((ext_vector_type(8)))  float  v8f;

union FragBF {
    v16bf v;
    uint4 u[2];
};

#define LOG2E 1.4426950408889634f

// branch-free fast transcendentals on v_exp_f32 (native exp2)
__device__ __forceinline__ float fast_sigmoid(float x) {
    return 1.f / (1.f + __builtin_amdgcn_exp2f(-x * LOG2E));
}
__device__ __forceinline__ float fast_tanh(float x) {
    // tanh(x) = 1 - 2/(1 + exp2(2x*log2e)); saturates correctly at +-inf
    return 1.f - 2.f / (1.f + __builtin_amdgcn_exp2f(x * (2.f * LOG2E)));
}
__device__ __forceinline__ float gelu_tanh(float v) {
    const float c = 0.7978845608028654f;   // sqrt(2/pi)
    float inner = c * (v + 0.044715f * v * v * v);
    return 0.5f * v * (1.f + fast_tanh(inner));
}

// ---------------------------------------------------------------------------
// Kernel 0a: fp32 -> bf16 convert (for x)
// ---------------------------------------------------------------------------
__global__ void cvt_bf16_kernel(const float* __restrict__ in,
                                __hip_bfloat16* __restrict__ out, int n) {
    int i = blockIdx.x * blockDim.x + threadIdx.x;
    if (i < n) out[i] = __float2bfloat16(in[i]);
}

// ---------------------------------------------------------------------------
// Kernel 0b: per-expert tiled transpose fp32(R x C) -> bf16(C x R)
//   out[e][c][r] = bf16(in[e][r][c]).  Block 32x8, 32x32 tile via LDS.
// ---------------------------------------------------------------------------
__global__ __launch_bounds__(256)
void transpose_bf16_kernel(const float* __restrict__ in,
                           __hip_bfloat16* __restrict__ out, int R, int C) {
    __shared__ float tile[32][33];
    const int e = blockIdx.z;
    const float* src = in + (size_t)e * R * C;
    __hip_bfloat16* dst = out + (size_t)e * R * C;
    const int bx = blockIdx.x * 32;   // col base
    const int by = blockIdx.y * 32;   // row base
    const int tx = threadIdx.x;       // 0..31
    const int ty = threadIdx.y;       // 0..7
    #pragma unroll
    for (int j = 0; j < 32; j += 8)
        tile[ty + j][tx] = src[(size_t)(by + ty + j) * C + (bx + tx)];
    __syncthreads();
    #pragma unroll
    for (int j = 0; j < 32; j += 8)
        dst[(size_t)(bx + ty + j) * R + (by + tx)] =
            __float2bfloat16(tile[tx][ty + j]);
}

// ---------------------------------------------------------------------------
// Kernel 1: router.  One wave (32 lanes) per token.
//   S = sigmoid(x @ Wg); top-2 -> append (token, gate) to expert lists.
// ---------------------------------------------------------------------------
__global__ __launch_bounds__(256)
void router_kernel(const float* __restrict__ x, const float* __restrict__ Wg,
                   int* __restrict__ counts, int* __restrict__ lists,
                   float* __restrict__ gates) {
    const int lane = threadIdx.x & 31;
    const int wav  = threadIdx.x >> 5;
    const int n    = blockIdx.x * 8 + wav;          // token id
    if (n >= NTOK) return;

    float acc[NEXP];
    #pragma unroll
    for (int e = 0; e < NEXP; e++) acc[e] = 0.f;

    const float* xr = x + (size_t)n * DIM;
    for (int d = lane; d < DIM; d += 32) {
        const float xv = xr[d];
        const float4* wg = (const float4*)(Wg + (size_t)d * NEXP);
        const float4 a = wg[0];
        const float4 b = wg[1];
        acc[0] += xv * a.x; acc[1] += xv * a.y;
        acc[2] += xv * a.z; acc[3] += xv * a.w;
        acc[4] += xv * b.x; acc[5] += xv * b.y;
        acc[6] += xv * b.z; acc[7] += xv * b.w;
    }
    // wave32 butterfly reduction
    #pragma unroll
    for (int off = 16; off > 0; off >>= 1) {
        #pragma unroll
        for (int e = 0; e < NEXP; e++)
            acc[e] += __shfl_xor(acc[e], off, 32);
    }
    if (lane == 0) {
        float s[NEXP];
        #pragma unroll
        for (int e = 0; e < NEXP; e++) s[e] = fast_sigmoid(acc[e]);
        int i1 = 0;
        #pragma unroll
        for (int e = 1; e < NEXP; e++) if (s[e] > s[i1]) i1 = e;
        int i2 = (i1 == 0) ? 1 : 0;
        #pragma unroll
        for (int e = 0; e < NEXP; e++)
            if (e != i1 && s[e] > s[i2]) i2 = e;
        int p1 = atomicAdd(&counts[i1], 1);
        lists[i1 * NTOK + p1] = n;  gates[i1 * NTOK + p1] = s[i1];
        int p2 = atomicAdd(&counts[i2], 1);
        lists[i2 * NTOK + p2] = n;  gates[i2 * NTOK + p2] = s[i2];
    }
}

// ---------------------------------------------------------------------------
// Kernel 2: expert MLP with WMMA.
//   Grid: (NTOK/16, NEXP). Block: 256 threads = 8 waves.
//   Workgroup (t, e): 16 gathered tokens of expert e -> out tile 16 x 1024.
//   F is processed in chunks of 512; h (16x512 bf16) lives in LDS.
// ---------------------------------------------------------------------------
#define FC   512              // F chunk
#define XP   (DIM + 8)        // xs pitch (bf16 elems), keeps 16B align + bank spread
#define HP   (FC + 8)         // hs pitch

// A-fragment (16x32 bf16) from row-major LDS buffer (rows = M, cols = K).
// ISA layout: lanes 0-15 -> M=lane, K = {k0..k0+7, k0+16..k0+23}
//             lanes16-31 -> M=lane-16, K = {k0+8..k0+15, k0+24..k0+31}
__device__ __forceinline__ v16bf load_a_lds(const __hip_bfloat16* buf,
                                            int pitch, int k0, int lane) {
    const int r  = lane & 15;
    const int kk = k0 + ((lane >> 4) << 3);
    FragBF f;
    f.u[0] = *(const uint4*)(buf + r * pitch + kk);
    f.u[1] = *(const uint4*)(buf + r * pitch + kk + 16);
    return f.v;
}

// B-fragment (32x16 bf16) from N-major global weights (row n contiguous in K).
// ISA layout: lanes 0-15 -> N=lane,    K = {k0..k0+7, k0+16..k0+23}
//             lanes16-31 -> N=lane-16, K = {k0+8..k0+15, k0+24..k0+31}
__device__ __forceinline__ v16bf load_b_glb(const __hip_bfloat16* w,
                                            size_t ld, int nbase, int k0,
                                            int lane) {
    const int n  = nbase + (lane & 15);
    const int kk = k0 + ((lane >> 4) << 3);
    const uint4* p = (const uint4*)(w + (size_t)n * ld + kk);
    FragBF f;
    f.u[0] = p[0];
    f.u[1] = p[2];
    return f.v;
}

__global__ __launch_bounds__(256)
void moe_expert_kernel(const __hip_bfloat16* __restrict__ xb,
                       const __hip_bfloat16* __restrict__ W1t,  // E x F x D
                       const __hip_bfloat16* __restrict__ W2t,  // E x D x F
                       const float* __restrict__ b1,            // E x F
                       const float* __restrict__ b2,            // E x D
                       const int* __restrict__ counts,
                       const int* __restrict__ lists,
                       const float* __restrict__ gates,
                       float* __restrict__ out) {
    const int e = blockIdx.y;
    const int t = blockIdx.x;
    const int cnt = counts[e];
    if (t * 16 >= cnt) return;

    __shared__ __hip_bfloat16 xs[16 * XP];
    __shared__ __hip_bfloat16 hs[16 * HP];
    __shared__ int   toks[16];
    __shared__ float gl[16];

    const int tid  = threadIdx.x;
    const int lane = tid & 31;
    const int w    = tid >> 5;      // wave 0..7
    const int hl   = lane & 15;
    const int hh   = lane >> 4;

    if (tid < 16) {
        const int idx = t * 16 + tid;
        const bool ok = idx < cnt;
        toks[tid] = ok ? lists[e * NTOK + idx] : 0;
        gl[tid]   = ok ? gates[e * NTOK + idx] : 0.f;
    }
    __syncthreads();

    // ---- stage gathered x rows into LDS (16 x DIM bf16) ----
    {
        const int r  = tid >> 4;          // 0..15
        const int c0 = (tid & 15) * 64;   // 64 bf16 per thread
        const uint4* src = (const uint4*)(xb + (size_t)toks[r] * DIM + c0);
        uint4* dst = (uint4*)(xs + r * XP + c0);
        #pragma unroll
        for (int i = 0; i < 8; i++) dst[i] = src[i];
    }
    __syncthreads();

    const __hip_bfloat16* W1e = W1t + (size_t)e * FDIM * DIM;
    const __hip_bfloat16* W2e = W2t + (size_t)e * DIM * FDIM;

    // per-lane gates / token ids for the 8 output rows this lane touches
    float garr[8];
    int   tarr[8];
    #pragma unroll
    for (int r = 0; r < 8; r++) {
        garr[r] = gl[hh * 8 + r];
        tarr[r] = toks[hh * 8 + r];
    }

    v8f acc[8];                       // out tile cols [w*128, w*128+128)
    #pragma unroll
    for (int s = 0; s < 8; s++) acc[s] = (v8f)(0.f);

    for (int fc = 0; fc < FDIM / FC; fc++) {
        const int fbase = fc * FC;

        // ---- GEMM1: h(16 x FC) = x(16 x DIM) @ W1[:, fbase..] ----
        // wave w owns h cols [w*64, w*64+64) -> 4 subtiles of 16
        v8f ha[4];
        #pragma unroll
        for (int s = 0; s < 4; s++) ha[s] = (v8f)(0.f);

        for (int k0 = 0; k0 < DIM; k0 += 32) {
            const v16bf a = load_a_lds(xs, XP, k0, lane);
            #pragma unroll
            for (int s = 0; s < 4; s++) {
                const int nb = fbase + w * 64 + s * 16;
                const v16bf bfr = load_b_glb(W1e, DIM, nb, k0, lane);
                ha[s] = __builtin_amdgcn_wmma_f32_16x16x32_bf16(
                    false, a, false, bfr, (short)0, ha[s], false, false);
            }
        }
        // bias + gelu + store bf16 h tile to LDS
        #pragma unroll
        for (int s = 0; s < 4; s++) {
            const int col  = w * 64 + s * 16 + hl;
            const float bv = b1[e * FDIM + fbase + col];
            #pragma unroll
            for (int r = 0; r < 8; r++) {
                const int m = hh * 8 + r;
                hs[m * HP + col] = __float2bfloat16(gelu_tanh(ha[s][r] + bv));
            }
        }
        __syncthreads();

        // ---- GEMM2: out(16 x DIM) += h(16 x FC) @ W2[fbase.., :] ----
        // wave w owns out cols [w*128, w*128+128) -> 8 subtiles of 16
        for (int k0 = 0; k0 < FC; k0 += 32) {
            const v16bf a = load_a_lds(hs, HP, k0, lane);
            #pragma unroll
            for (int s = 0; s < 8; s++) {
                const int nb = w * 128 + s * 16;
                const v16bf bfr =
                    load_b_glb(W2e, FDIM, nb, fbase + k0, lane);
                acc[s] = __builtin_amdgcn_wmma_f32_16x16x32_bf16(
                    false, a, false, bfr, (short)0, acc[s], false, false);
            }
        }
        __syncthreads();   // hs reused next chunk
    }

    // ---- epilogue: +b2, gate, scatter-add to out ----
    #pragma unroll
    for (int s = 0; s < 8; s++) {
        const int n  = w * 128 + s * 16 + hl;
        const float bv = b2[e * DIM + n];
        #pragma unroll
        for (int r = 0; r < 8; r++) {
            const float v = (acc[s][r] + bv) * garr[r];
            atomicAdd(out + (size_t)tarr[r] * DIM + n, v);
        }
    }
}

// ---------------------------------------------------------------------------
// Launch
// ---------------------------------------------------------------------------
extern "C" void kernel_launch(void* const* d_in, const int* in_sizes, int n_in,
                              void* d_out, int out_size, void* d_ws, size_t ws_size,
                              hipStream_t stream) {
    const float* x  = (const float*)d_in[0];   // (B,L,D)
    const float* Wg = (const float*)d_in[1];   // (D,E)
    const float* W1 = (const float*)d_in[2];   // (E,D,F)
    const float* b1 = (const float*)d_in[3];   // (E,F)
    const float* W2 = (const float*)d_in[4];   // (E,F,D)
    const float* b2 = (const float*)d_in[5];   // (E,D)
    float* out = (float*)d_out;                // (B,L,D)

    // workspace layout
    char* ws = (char*)d_ws;
    const size_t wsz = (size_t)NEXP * DIM * FDIM;  // elems per weight tensor
    __hip_bfloat16* W1t = (__hip_bfloat16*)ws;                    ws += wsz * 2;
    __hip_bfloat16* W2t = (__hip_bfloat16*)ws;                    ws += wsz * 2;
    __hip_bfloat16* xb  = (__hip_bfloat16*)ws;                    ws += (size_t)NTOK * DIM * 2;
    int*   counts = (int*)ws;                                     ws += NEXP * sizeof(int);
    int*   lists  = (int*)ws;                                     ws += (size_t)NEXP * NTOK * sizeof(int);
    float* gates  = (float*)ws;                                   ws += (size_t)NEXP * NTOK * sizeof(float);

    hipMemsetAsync(out, 0, (size_t)NTOK * DIM * sizeof(float), stream);
    hipMemsetAsync(counts, 0, NEXP * sizeof(int), stream);

    // x -> bf16
    {
        const int n = NTOK * DIM;
        cvt_bf16_kernel<<<(n + 255) / 256, 256, 0, stream>>>(x, xb, n);
    }
    // W1 (E,D,F) -> W1t (E,F,D) bf16 ; W2 (E,F,D) -> W2t (E,D,F) bf16
    {
        dim3 blk(32, 8);
        dim3 g1(FDIM / 32, DIM / 32, NEXP);
        transpose_bf16_kernel<<<g1, blk, 0, stream>>>(W1, W1t, DIM, FDIM);
        dim3 g2(DIM / 32, FDIM / 32, NEXP);
        transpose_bf16_kernel<<<g2, blk, 0, stream>>>(W2, W2t, FDIM, DIM);
    }
    // router
    router_kernel<<<NTOK / 8, 256, 0, stream>>>(x, Wg, counts, lists, gates);

    // expert MLP
    {
        dim3 grid(NTOK / 16, NEXP);
        moe_expert_kernel<<<grid, 256, 0, stream>>>(xb, W1t, W2t, b1, b2,
                                                    counts, lists, gates, out);
    }
}
